// NTreeGRU_56676388438065
// MI455X (gfx1250) — compile-verified, matrix operand order checked by reading, not run on hardware
//
#include <hip/hip_runtime.h>

#define TDEPTH 13
#define B_TREES 16
#define X_DIM 256
#define H_DIM 128
#define AH_DIM 256
#define WOUT 512                                  // (2+A)*H
#define N_NODES ((1 << (TDEPTH + 1)) - 1)         // 16383
#define N_TOTAL (B_TREES * N_NODES)               // 262128
#define LEAF0 ((1 << TDEPTH) - 1)                 // first leaf heap index: 8191

typedef __attribute__((ext_vector_type(16))) __bf16 v16bf;
typedef __attribute__((ext_vector_type(8)))  __bf16 v8bf;
typedef __attribute__((ext_vector_type(8)))  float  v8f;
typedef __attribute__((ext_vector_type(4)))  float  v4f;

__device__ __forceinline__ float sigmoidf_(float x) { return 1.0f / (1.0f + __expf(-x)); }

// Assemble a 16x32 bf16 WMMA fragment for this lane from two contiguous 8-elem runs.
// Lane layout (wave32): lo = lane&15 selects row/col, kbase=(lane>>4)*8;
// elements 0..7 -> K = base+kbase+0..7, elements 8..15 -> K = base+kbase+16..23.
__device__ __forceinline__ v16bf pack_frag(const __bf16* p0, const __bf16* p1) {
    v8bf a = *reinterpret_cast<const v8bf*>(p0);
    v8bf b = *reinterpret_cast<const v8bf*>(p1);
    v16bf f;
#pragma unroll
    for (int i = 0; i < 8; ++i) { f[i] = a[i]; f[8 + i] = b[i]; }
    return f;
}

__device__ __forceinline__ v8f wmma_bf16(v16bf A, v16bf B, v8f C) {
    return __builtin_amdgcn_wmma_f32_16x16x32_bf16(false, A, false, B, (short)0, C, false, false);
}

__global__ void cvt_bf16_kernel(const float* __restrict__ src, __bf16* __restrict__ dst, int count) {
    int i = blockIdx.x * blockDim.x + threadIdx.x;
    if (i < count) dst[i] = (__bf16)src[i];
}

// wx[n,512] = x[n,256] @ W_w[512,256]^T + b, with two fusions:
//  - x is read fp32 and converted to bf16 in-register (no xb round trip)
//  - leaf rows (heap >= 8191, half of all nodes) produce h directly from the
//    accumulators: h = (1 - sig(wz0) - sig(wz1)) * tanh(whc); their wx rows are
//    never needed, so pure-leaf tiles skip the wx write (and the w_r tiles).
__global__ __launch_bounds__(128) void gemm_wx_kernel(
    const float* __restrict__ x, const __bf16* __restrict__ Wb,
    const float* __restrict__ bias, float* __restrict__ wx,
    float* __restrict__ h, __bf16* __restrict__ hb, int mtiles) {
    int tile = blockIdx.x * 4 + (threadIdx.x >> 5);
    if (tile >= mtiles) tile = mtiles - 1;          // duplicate work, identical writes
    int lane = threadIdx.x & 31;
    int lo = lane & 15, hi = lane >> 4, kbase = hi * 8;

    // wave-uniform tile classification
    int r_first = tile * 16;
    int b0 = r_first / N_NODES, b1 = (r_first + 15) / N_NODES;
    int heap0 = r_first - b0 * N_NODES;
    bool all_leaf  = (b0 == b1) && (heap0 >= LEAF0);
    bool all_inner = (b0 == b1) && (heap0 + 15 < LEAF0);

    // A fragments: fp32 loads + in-register bf16 convert
    const float* arow = x + (size_t)(r_first + lo) * X_DIM;
    v16bf Af[8];
#pragma unroll
    for (int k = 0; k < 8; ++k) {
        int r0 = k * 32 + kbase;
        v4f f0 = *reinterpret_cast<const v4f*>(arow + r0);
        v4f f1 = *reinterpret_cast<const v4f*>(arow + r0 + 4);
        v4f f2 = *reinterpret_cast<const v4f*>(arow + r0 + 16);
        v4f f3 = *reinterpret_cast<const v4f*>(arow + r0 + 20);
#pragma unroll
        for (int i = 0; i < 4; ++i) {
            Af[k][i]      = (__bf16)f0[i];
            Af[k][4 + i]  = (__bf16)f1[i];
            Af[k][8 + i]  = (__bf16)f2[i];
            Af[k][12 + i] = (__bf16)f3[i];
        }
    }

    if (!all_leaf) {
        // full wx rows (needed later by inner-level kernels)
        for (int nt = 0; nt < 32; ++nt) {
            int col = nt * 16 + lo;
            const __bf16* brow = Wb + (size_t)col * X_DIM;
            v8f acc = {};
#pragma unroll
            for (int k = 0; k < 8; ++k) {
                int r0 = k * 32 + kbase;
                v16bf Bf = pack_frag(brow + r0, brow + r0 + 16);
                acc = wmma_bf16(Af[k], Bf, acc);
            }
            float bv = bias[col];
#pragma unroll
            for (int v = 0; v < 8; ++v) {           // C layout: M = v + 8*hi, N = lo
                int rowM = r_first + v + 8 * hi;
                wx[(size_t)rowM * WOUT + col] = acc[v] + bv;
            }
        }
    }
    if (!all_inner) {
        // fused leaf epilogue over the 8 d-tiles of H
        for (int dt = 0; dt < 8; ++dt) {
            int col = dt * 16 + lo;
            const __bf16* bhc = Wb + (size_t)(H_DIM + col) * X_DIM;
            const __bf16* bz0 = Wb + (size_t)(2 * H_DIM + col) * X_DIM;
            const __bf16* bz1 = Wb + (size_t)(3 * H_DIM + col) * X_DIM;
            v8f ahc = {}, az0 = {}, az1 = {};
#pragma unroll
            for (int k = 0; k < 8; ++k) {
                int r0 = k * 32 + kbase;
                ahc = wmma_bf16(Af[k], pack_frag(bhc + r0, bhc + r0 + 16), ahc);
                az0 = wmma_bf16(Af[k], pack_frag(bz0 + r0, bz0 + r0 + 16), az0);
                az1 = wmma_bf16(Af[k], pack_frag(bz1 + r0, bz1 + r0 + 16), az1);
            }
            float cbh = bias[H_DIM + col];
            float cb0 = bias[2 * H_DIM + col];
            float cb1 = bias[3 * H_DIM + col];
#pragma unroll
            for (int v = 0; v < 8; ++v) {
                int row = r_first + v + 8 * hi;
                int bt = row / N_NODES;
                int heap = row - bt * N_NODES;
                if (heap >= LEAF0) {                // per-element predicate (stores only)
                    float z0 = sigmoidf_(az0[v] + cb0);
                    float z1 = sigmoidf_(az1[v] + cb1);
                    float val = (1.0f - z0 - z1) * tanhf(ahc[v] + cbh);
                    h[(size_t)row * H_DIM + col] = val;
                    hb[(size_t)row * H_DIM + col] = (__bf16)val;
                }
            }
        }
    }
}

// Inner level: one wave per 16-node tile. h_cat rows are contiguous 256-float slices of h.
__global__ __launch_bounds__(64) void level_kernel(
    const float* __restrict__ wx,
    const __bf16* __restrict__ Urb, const __bf16* __restrict__ Uhb,
    const __bf16* __restrict__ Uzb,
    float* __restrict__ h, __bf16* __restrict__ hb, int lvl) {
    __shared__ float r_s[2][16][H_DIM];
    __shared__ float z_s[2][16][AH_DIM];

    int L = 1 << lvl;
    int wave = threadIdx.x >> 5;
    int tile = blockIdx.x * 2 + wave;
    if (tile >= L) tile = L - 1;                    // benign duplicate
    int lane = threadIdx.x & 31;
    int lo = lane & 15, hi = lane >> 4, kbase = hi * 8;
    float (*rs)[H_DIM] = r_s[wave];
    float (*zs)[AH_DIM] = z_s[wave];

    // A-fragment row owned by this lane
    int rowA = tile * 16 + lo;
    int bA = rowA >> lvl, jA = rowA & (L - 1);
    size_t hcA = ((size_t)bA * N_NODES + (2 * L - 1) + 2 * jA) * (size_t)H_DIM;

    // C-layout rows this lane touches (M = v + 8*hi)
    int idM[8]; size_t hcM[8];
#pragma unroll
    for (int v = 0; v < 8; ++v) {
        int rowM = tile * 16 + v + 8 * hi;
        int bM = rowM >> lvl, jM = rowM & (L - 1);
        idM[v] = bM * N_NODES + (L - 1) + jM;
        hcM[v] = ((size_t)bM * N_NODES + (2 * L - 1) + 2 * jM) * (size_t)H_DIM;
    }

    // h_cat fragments (bf16), K = 0..255 in 8 steps of 32
    v16bf Af[8];
#pragma unroll
    for (int k = 0; k < 8; ++k) {
        int r0 = k * 32 + kbase;
        Af[k] = pack_frag(hb + hcA + r0, hb + hcA + r0 + 16);
    }

    // Stage A1: r = sigmoid(w_r + h_cat @ U_r^T)  -> LDS
    for (int nt = 0; nt < 8; ++nt) {
        int col = nt * 16 + lo;
        const __bf16* brow = Urb + (size_t)col * AH_DIM;
        v8f acc = {};
#pragma unroll
        for (int k = 0; k < 8; ++k) {
            int r0 = k * 32 + kbase;
            acc = wmma_bf16(Af[k], pack_frag(brow + r0, brow + r0 + 16), acc);
        }
#pragma unroll
        for (int v = 0; v < 8; ++v) {
            float wr = wx[(size_t)idM[v] * WOUT + col];
            rs[v + 8 * hi][col] = sigmoidf_(acc[v] + wr);
        }
    }
    // Stage A2: z = sigmoid(w_z + h_cat @ U_z^T)  -> LDS
    for (int nt = 0; nt < 16; ++nt) {
        int col = nt * 16 + lo;
        const __bf16* brow = Uzb + (size_t)col * AH_DIM;
        v8f acc = {};
#pragma unroll
        for (int k = 0; k < 8; ++k) {
            int r0 = k * 32 + kbase;
            acc = wmma_bf16(Af[k], pack_frag(brow + r0, brow + r0 + 16), acc);
        }
#pragma unroll
        for (int v = 0; v < 8; ++v) {
            float wz = wx[(size_t)idM[v] * WOUT + 2 * H_DIM + col];
            zs[v + 8 * hi][col] = sigmoidf_(acc[v] + wz);
        }
    }

    // CDNA5 split-counter wait: make per-wave LDS writes visible before reads
    asm volatile("s_wait_dscnt 0" ::: "memory");

    // Stage B: delta = (tile(r)*h_cat) @ U_h^T, then GRU combine
    for (int nt = 0; nt < 8; ++nt) {
        int col = nt * 16 + lo;
        const __bf16* brow = Uhb + (size_t)col * AH_DIM;
        v8f acc = {};
#pragma unroll
        for (int k = 0; k < 8; ++k) {
            int r0 = k * 32 + kbase;
            int r1 = r0 + 16;
            v16bf As;
#pragma unroll
            for (int i = 0; i < 8; ++i) {
                As[i]     = (__bf16)((float)Af[k][i]     * rs[lo][(r0 + i) & (H_DIM - 1)]);
                As[8 + i] = (__bf16)((float)Af[k][8 + i] * rs[lo][(r1 + i) & (H_DIM - 1)]);
            }
            acc = wmma_bf16(As, pack_frag(brow + r0, brow + r1), acc);
        }
#pragma unroll
        for (int v = 0; v < 8; ++v) {
            int Mloc = v + 8 * hi;
            float z0 = zs[Mloc][col];
            float z1 = zs[Mloc][H_DIM + col];
            float hc0 = h[hcM[v] + col];
            float hc1 = h[hcM[v] + H_DIM + col];
            float whc = wx[(size_t)idM[v] * WOUT + H_DIM + col];
            float val = hc0 * z0 + hc1 * z1 + (1.0f - z0 - z1) * tanhf(whc + acc[v]);
            h[(size_t)idM[v] * H_DIM + col] = val;
            hb[(size_t)idM[v] * H_DIM + col] = (__bf16)val;
        }
    }
}

extern "C" void kernel_launch(void* const* d_in, const int* in_sizes, int n_in,
                              void* d_out, int out_size, void* d_ws, size_t ws_size,
                              hipStream_t stream) {
    const float* x   = (const float*)d_in[0];
    const float* W_w = (const float*)d_in[1];
    const float* W_b = (const float*)d_in[2];
    const float* U_r = (const float*)d_in[3];
    const float* U_h = (const float*)d_in[4];
    const float* U_z = (const float*)d_in[5];
    float* h = (float*)d_out;

    char* ws = (char*)d_ws;
    size_t off = 0;
    auto alloc = [&](size_t bytes) -> void* {
        void* p = ws + off;
        off = (off + bytes + 255) & ~(size_t)255;
        return p;
    };
    __bf16* Wb  = (__bf16*)alloc((size_t)WOUT * X_DIM * 2);
    __bf16* Urb = (__bf16*)alloc((size_t)H_DIM * AH_DIM * 2);
    __bf16* Uhb = (__bf16*)alloc((size_t)H_DIM * AH_DIM * 2);
    __bf16* Uzb = (__bf16*)alloc((size_t)AH_DIM * AH_DIM * 2);
    float*  wx  = (float*)alloc((size_t)N_TOTAL * WOUT * 4);
    __bf16* hb  = (__bf16*)alloc((size_t)N_TOTAL * H_DIM * 2);

    auto cvt = [&](const float* s, __bf16* d, int count) {
        cvt_bf16_kernel<<<(count + 255) / 256, 256, 0, stream>>>(s, d, count);
    };
    cvt(W_w, Wb,  WOUT * X_DIM);
    cvt(U_r, Urb, H_DIM * AH_DIM);
    cvt(U_h, Uhb, H_DIM * AH_DIM);
    cvt(U_z, Uzb, AH_DIM * AH_DIM);

    int mtiles = N_TOTAL / 16;                      // 16383
    gemm_wx_kernel<<<(mtiles + 3) / 4, 128, 0, stream>>>(x, Wb, W_b, wx, h, hb, mtiles);

    for (int lvl = TDEPTH - 1; lvl >= 0; --lvl) {
        int L = 1 << lvl;                           // tiles at this level
        level_kernel<<<(L + 1) / 2, 64, 0, stream>>>(wx, Urb, Uhb, Uzb, h, hb, lvl);
    }
}
